// Decoder_39968965656603
// MI455X (gfx1250) — matrix operand order repaired and checked by reading, not compiled, not run on hardware
//
#include <hip/hip_runtime.h>
#include <hip/hip_bf16.h>

// ---------------------------------------------------------------------------
// Model dims
// ---------------------------------------------------------------------------
#define Bk      256
#define Lk      128
#define Hk      512
#define NHSA    8
#define NHDEC   2
#define FFk     300
#define SLOTk   130
#define INTENTk 30
#define BLk     (Bk * Lk)          // 32768 tokens

typedef _Float16 f16;
typedef __attribute__((ext_vector_type(16))) _Float16 v16h;
typedef __attribute__((ext_vector_type(8)))  float    v8f;
typedef long long ll;

#define NEG_INF (-__builtin_inff())

__device__ __forceinline__ unsigned pack2h(float x, float y) {
    union { f16 h[2]; unsigned u; } t;
    t.h[0] = (f16)x;
    t.h[1] = (f16)y;
    return t.u;
}

// ---------------------------------------------------------------------------
// Generic batched WMMA GEMM:
//   C[z] = alpha * A[z](MxK) * B[z](KxN or NxK if transB) + bias, opt ReLU
// Batch z -> b = z/nh, h = z%nh; per-operand offset = b*s?b + h*s?h.
// Block: 256 threads (8 waves), tile 64(M) x 128(N), BK=32.
// Each wave computes a 32x32 subtile = 2x2 v_wmma_f32_16x16x32_f16.
//
// Staging: interior tiles use GLOBAL_LOAD_ASYNC_TO_LDS_{B128,B32} (ASYNCcnt)
// into double-buffered f32 LDS tiles; edge tiles fall back to branch-free
// clamped sync loads. f32->f16 conversion happens during fragment assembly
// (co-executes with WMMA). One barrier per K-step.
// ---------------------------------------------------------------------------
#define ASTR 36    // f32 per A row   (32 + pad), 144B: 16B-aligned rows
#define BSTR 132   // f32 per B k-row (128 + pad), 528B: 16B-aligned rows

__global__ __launch_bounds__(256) void wmma_gemm_kernel(
    const float* __restrict__ A, const float* __restrict__ Bw,
    const float* __restrict__ bias, float* __restrict__ C,
    int M, int N, int K,
    ll sAb, ll sAh, int lda,
    ll sBb, ll sBh, int ldb, int transB,
    ll sCb, ll sCh, int ldc,
    int nh, float alpha, int relu)
{
    __shared__ __align__(16) float Asf[2][64 * ASTR];
    __shared__ __align__(16) float Bsf[2][32 * BSTR];

    const int z  = blockIdx.z;
    const int bb = z / nh;
    const int hh = z - bb * nh;
    const float* __restrict__ Ab = A  + bb * sAb + hh * sAh;
    const float* __restrict__ Bb = Bw + bb * sBb + hh * sBh;
    float*       __restrict__ Cb = C  + bb * sCb + hh * sCh;

    const int m0   = blockIdx.y * 64;
    const int n0   = blockIdx.x * 128;
    const int tid  = threadIdx.x;
    const int lane = tid & 31;
    const int wave = tid >> 5;
    const int wr   = wave >> 2;   // 0..1 -> 32-row band
    const int wc   = wave & 3;    // 0..3 -> 32-col band

    v8f acc00 = {}; v8f acc01 = {}; v8f acc10 = {}; v8f acc11 = {};

    const int hiA  = (lane >= 16) ? 8 : 0;    // A-frag f32-pair offset
    const int koff = (lane >= 16) ? 16 : 0;   // B-frag K offset
    const int mA0  = wr * 32 + (lane & 15);
    const int mA1  = mA0 + 16;
    const int nB0  = wc * 32 + (lane & 15);
    const int nB1  = nB0 + 16;

    const bool mEdge = (m0 + 64 > M);
    const bool nEdge = (n0 + 128 > N);
    const bool aAsync  = ((lda & 3) == 0);
    const bool bAsyn128 = (!transB) && ((ldb & 3) == 0);

    // ---- sync (fallback) staging: branch-free clamped loads ----
    auto syncA = [&](int k0, float* Ad) {
#pragma unroll
        for (int i = 0; i < 8; ++i) {
            int mm = (tid >> 5) + i * 8;
            int kk = tid & 31;
            int gm = m0 + mm, gk = k0 + kk;
            int gmc = gm < M ? gm : M - 1;
            int gkc = gk < K ? gk : K - 1;
            float v = Ab[(ll)gmc * lda + gkc];
            Ad[mm * ASTR + kk] = (gm < M && gk < K) ? v : 0.f;
        }
    };
    auto syncB = [&](int k0, float* Bd) {
#pragma unroll
        for (int i = 0; i < 16; ++i) {
            int kk = (tid >> 7) + i * 2;
            int nn = tid & 127;
            int gk = k0 + kk, gn = n0 + nn;
            int gkc = gk < K ? gk : K - 1;
            int gnc = gn < N ? gn : N - 1;
            ll off = transB ? ((ll)gnc * ldb + gkc) : ((ll)gkc * ldb + gnc);
            float v = Bb[off];
            Bd[kk * BSTR + nn] = (gk < K && gn < N) ? v : 0.f;
        }
    };

    // ---- stage one K-tile into LDS buffer `buf` ----
    auto stage = [&](int k0, int buf) {
        float* Ad = Asf[buf];
        float* Bd = Bsf[buf];
        bool interior = (k0 + 32 <= K) && !nEdge && !mEdge;
        if (interior) {
            if (aAsync) {
#pragma unroll
                for (int i = 0; i < 2; ++i) {
                    int mm = (tid >> 3) + i * 32;
                    int kq = (tid & 7) * 4;
                    const float* g = Ab + (ll)(m0 + mm) * lda + (k0 + kq);
                    unsigned lo = (unsigned)(unsigned long long)(Ad + mm * ASTR + kq);
                    asm volatile("global_load_async_to_lds_b128 %0, %1, off"
                                 :: "v"(lo), "v"(g) : "memory");
                }
            } else {
                syncA(k0, Ad);
            }
            if (bAsyn128) {
#pragma unroll
                for (int i = 0; i < 4; ++i) {
                    int kk = (tid >> 5) + i * 8;
                    int nq = (tid & 31) * 4;
                    const float* g = Bb + (ll)(k0 + kk) * ldb + (n0 + nq);
                    unsigned lo = (unsigned)(unsigned long long)(Bd + kk * BSTR + nq);
                    asm volatile("global_load_async_to_lds_b128 %0, %1, off"
                                 :: "v"(lo), "v"(g) : "memory");
                }
            } else {
#pragma unroll
                for (int i = 0; i < 16; ++i) {
                    int kk = (tid >> 7) + i * 2;
                    int nn = tid & 127;
                    ll off = transB ? ((ll)(n0 + nn) * ldb + (k0 + kk))
                                    : ((ll)(k0 + kk) * ldb + (n0 + nn));
                    const float* g = Bb + off;
                    unsigned lo = (unsigned)(unsigned long long)(Bd + kk * BSTR + nn);
                    asm volatile("global_load_async_to_lds_b32 %0, %1, off"
                                 :: "v"(lo), "v"(g) : "memory");
                }
            }
        } else {
            syncA(k0, Ad);
            syncB(k0, Bd);
        }
    };

    // ---- fragment assembly (f32 LDS -> f16 fragments) + 4 WMMA ----
    auto compute = [&](int buf) {
        const float* Ac = Asf[buf];
        const float* Bc = Bsf[buf];
        union Frag { v16h v; unsigned u[8]; } a0, a1, b0, b1;
        const float2* pa0 = (const float2*)(Ac + mA0 * ASTR);
        const float2* pa1 = (const float2*)(Ac + mA1 * ASTR);
#pragma unroll
        for (int j = 0; j < 8; ++j) {
            int ja = j + hiA;                        // A: K pairs {0..15}(+16 hi)
            float2 t0 = pa0[ja];
            float2 t1 = pa1[ja];
            a0.u[j] = pack2h(t0.x, t0.y);
            a1.u[j] = pack2h(t1.x, t1.y);
            int k2 = 2 * j + koff;                   // B: K pairs {0..15}(+16 hi)
            b0.u[j] = pack2h(Bc[k2 * BSTR + nB0], Bc[(k2 + 1) * BSTR + nB0]);
            b1.u[j] = pack2h(Bc[k2 * BSTR + nB1], Bc[(k2 + 1) * BSTR + nB1]);
        }
        acc00 = __builtin_amdgcn_wmma_f32_16x16x32_f16(false, a0.v, false, b0.v, (short)0, acc00, false, false);
        acc01 = __builtin_amdgcn_wmma_f32_16x16x32_f16(false, a0.v, false, b1.v, (short)0, acc01, false, false);
        acc10 = __builtin_amdgcn_wmma_f32_16x16x32_f16(false, a1.v, false, b0.v, (short)0, acc10, false, false);
        acc11 = __builtin_amdgcn_wmma_f32_16x16x32_f16(false, a1.v, false, b1.v, (short)0, acc11, false, false);
    };

    // ---- pipelined main loop: one barrier per K-step ----
    stage(0, 0);
    int cur = 0;
    for (int k0 = 0; k0 < K; k0 += 32) {
        asm volatile("s_wait_asynccnt 0x0" ::: "memory");
        __syncthreads();
        if (k0 + 32 < K) stage(k0 + 32, cur ^ 1);
        if (k0 + 96 <= K && !nEdge && !mEdge) {   // prefetch 2 tiles ahead
            const float* pA = Ab + (ll)(m0 + (tid >> 5)) * lda + (k0 + 64 + (tid & 31));
            __builtin_prefetch(pA, 0, 3);
            ll poff = transB ? ((ll)(n0 + (tid & 127)) * ldb + (k0 + 64 + (tid >> 7)))
                             : ((ll)(k0 + 64 + (tid >> 7)) * ldb + (n0 + (tid & 127)));
            __builtin_prefetch(Bb + poff, 0, 3);
        }
        compute(cur);
        cur ^= 1;
    }

    // ---- epilogue: C/D layout -> lane covers (r + 8*hi, lane&15) ----
    const int rbase = (lane >= 16) ? 8 : 0;
    const int cn    = lane & 15;
    auto store_tile = [&](v8f acc, int tm, int tn) {
        int gmb = m0 + wr * 32 + tm * 16 + rbase;
        int gn  = n0 + wc * 32 + tn * 16 + cn;
        if (gn >= N) return;
        float bv = bias ? bias[gn] : 0.f;
#pragma unroll
        for (int r = 0; r < 8; ++r) {
            int gm = gmb + r;
            if (gm < M) {
                float v = alpha * acc[r] + bv;
                if (relu) v = v > 0.f ? v : 0.f;
                Cb[(ll)gm * ldc + gn] = v;
            }
        }
    };
    store_tile(acc00, 0, 0);
    store_tile(acc01, 0, 1);
    store_tile(acc10, 1, 0);
    store_tile(acc11, 1, 1);
}

// ---------------------------------------------------------------------------
// Embedding + positional encoding (PE indexed by BATCH per reference quirk)
// ---------------------------------------------------------------------------
__global__ void embed_pe_kernel(const int* __restrict__ ids,
                                const float* __restrict__ emb,
                                float* __restrict__ x)
{
    ll idx = (ll)blockIdx.x * blockDim.x + threadIdx.x;   // over BL*H
    if (idx >= (ll)BLk * Hk) return;
    int d = (int)(idx % Hk);
    ll  t = idx / Hk;
    int l = (int)(t % Lk);
    int b = (int)(t / Lk);
    int id = ids[b * Lk + l];
    float e = emb[(ll)id * Hk + d] * 22.627416997969522f;  // sqrt(512)
    int i2 = d & ~1;
    float freq = __expf(-(float)i2 * (9.210340371976184f / 512.0f)); // ln(1e4)
    float ang  = (float)b * freq;
    float pe   = (d & 1) ? __cosf(ang) : __sinf(ang);
    x[idx] = e + pe;
}

// ---------------------------------------------------------------------------
// Row softmax over L=128 (mode 0: none, 1: causal). One row per 128-thr block.
// ---------------------------------------------------------------------------
__global__ void softmax_kernel(float* __restrict__ s, int mode)
{
    __shared__ float red[128];
    int row = blockIdx.x;
    int c   = threadIdx.x;
    float* p = s + (ll)row * Lk;
    float v = p[c];
    if (mode == 1) {
        int q = row & (Lk - 1);
        if (c > q) v = NEG_INF;
    }
    red[c] = v; __syncthreads();
    for (int st = 64; st > 0; st >>= 1) {
        if (c < st) red[c] = fmaxf(red[c], red[c + st]);
        __syncthreads();
    }
    float mx = red[0]; __syncthreads();
    float e = __expf(v - mx);
    red[c] = e; __syncthreads();
    for (int st = 64; st > 0; st >>= 1) {
        if (c < st) red[c] += red[c + st];
        __syncthreads();
    }
    p[c] = e / red[0];
}

// ---------------------------------------------------------------------------
// Fused add + LayerNorm over H=512. mode 0: y=LN(x+res); mode 1: y=LN(x)+res.
// Safe for y aliasing x or res (row fully read before written).
// ---------------------------------------------------------------------------
__global__ __launch_bounds__(256) void add_ln_kernel(
    float* __restrict__ y, const float* __restrict__ x,
    const float* __restrict__ res,
    const float* __restrict__ g, const float* __restrict__ b, int mode)
{
    __shared__ float r1[256], r2[256];
    ll row = blockIdx.x;
    const float* xr = x   + row * Hk;
    const float* rr = res + row * Hk;
    float*       yr = y   + row * Hk;
    int t = threadIdx.x;
    float v0 = xr[t], v1 = xr[t + 256];
    float q0 = rr[t], q1 = rr[t + 256];
    if (mode == 0) { v0 += q0; v1 += q1; }
    r1[t] = v0 + v1;
    r2[t] = v0 * v0 + v1 * v1;
    __syncthreads();
    for (int st = 128; st > 0; st >>= 1) {
        if (t < st) { r1[t] += r1[t + st]; r2[t] += r2[t + st]; }
        __syncthreads();
    }
    float mean = r1[0] * (1.0f / Hk);
    float var  = r2[0] * (1.0f / Hk) - mean * mean;
    float inv  = rsqrtf(var + 1e-5f);
    float o0 = (v0 - mean) * inv * g[t]       + b[t];
    float o1 = (v1 - mean) * inv * g[t + 256] + b[t + 256];
    if (mode == 1) { o0 += q0; o1 += q1; }
    yr[t]       = o0;
    yr[t + 256] = o1;
}

// ---------------------------------------------------------------------------
// Masked mean pool over L. mask byte !=0 means PADDED (invalid).
// ---------------------------------------------------------------------------
__global__ void meanpool_kernel(const float* __restrict__ enc2,
                                const unsigned char* __restrict__ mask,
                                float* __restrict__ pooled)
{
    int b = blockIdx.x;
    int d = blockIdx.y * 256 + threadIdx.x;
    float s = 0.f, cnt = 0.f;
    for (int l = 0; l < Lk; ++l) {
        if (mask[b * Lk + l] == 0) {
            s += enc2[((ll)b * Lk + l) * Hk + d];
            cnt += 1.f;
        }
    }
    pooled[(ll)b * Hk + d] = s / fmaxf(cnt, 1.f);
}

// ---------------------------------------------------------------------------
// Ragged subtoken compaction: kept positions first (stable), zero tail.
// ---------------------------------------------------------------------------
__global__ void compact_kernel(const float* __restrict__ enc,
                               const int* __restrict__ m,
                               float* __restrict__ nt)
{
    __shared__ int pos[Lk];
    __shared__ int cnt;
    int b = blockIdx.x;
    if (threadIdx.x == 0) {
        int c = 0;
        for (int l = 0; l < Lk; ++l)
            if (m[b * Lk + l] != 0) pos[c++] = l;
        cnt = c;
    }
    __syncthreads();
    for (int j = 0; j < Lk; ++j) {
        int keep = (j < cnt);
        int src  = keep ? pos[j] : 0;
        for (int d = threadIdx.x; d < Hk; d += blockDim.x) {
            float v = keep ? enc[((ll)b * Lk + src) * Hk + d] : 0.f;
            nt[((ll)b * Lk + j) * Hk + d] = v;
        }
    }
}

// ---------------------------------------------------------------------------
// In-place log-softmax over n (<=256) columns; one row per 256-thr block.
// ---------------------------------------------------------------------------
__global__ __launch_bounds__(256) void log_softmax_kernel(float* __restrict__ s, int n)
{
    __shared__ float red[256];
    int row = blockIdx.x;
    int t   = threadIdx.x;
    float* p = s + (ll)row * n;
    float v = (t < n) ? p[t] : NEG_INF;
    red[t] = v; __syncthreads();
    for (int st = 128; st > 0; st >>= 1) {
        if (t < st) red[t] = fmaxf(red[t], red[t + st]);
        __syncthreads();
    }
    float mx = red[0]; __syncthreads();
    float e = (t < n) ? __expf(v - mx) : 0.f;
    red[t] = e; __syncthreads();
    for (int st = 128; st > 0; st >>= 1) {
        if (t < st) red[t] += red[t + st];
        __syncthreads();
    }
    float ls = __logf(red[0]);
    if (t < n) p[t] = v - mx - ls;
}

// ---------------------------------------------------------------------------
// Host orchestration
// ---------------------------------------------------------------------------
struct AttnW { const float *Wq,*bq,*Wk,*bk,*Wv,*bv,*Wo,*bo; };

static void launch_gemm(hipStream_t st,
                        const float* A, const float* Bw, const float* bias, float* C,
                        int M, int N, int K,
                        ll sAb, ll sAh, int lda,
                        ll sBb, ll sBh, int ldb, int transB,
                        ll sCb, ll sCh, int ldc,
                        int nh, int batches, float alpha, int relu)
{
    dim3 grid((N + 127) / 128, (M + 63) / 64, batches);
    wmma_gemm_kernel<<<grid, 256, 0, st>>>(A, Bw, bias, C, M, N, K,
                                           sAb, sAh, lda, sBb, sBh, ldb, transB,
                                           sCb, sCh, ldc, nh, alpha, relu);
}

// token-level GEMM: [BL x K] @ [K x N] + bias
static void tok_gemm(hipStream_t st, const float* A, const float* W, const float* bias,
                     float* C, int N, int K, int relu)
{
    launch_gemm(st, A, W, bias, C, BLk, N, K,
                0, 0, K, 0, 0, N, 0, 0, 0, N, 1, 1, 1.0f, relu);
}

extern "C" void kernel_launch(void* const* d_in, const int* in_sizes, int n_in,
                              void* d_out, int out_size, void* d_ws, size_t ws_size,
                              hipStream_t stream)
{
    (void)in_sizes; (void)n_in; (void)out_size; (void)ws_size;

    // ---- inputs (setup_inputs dict order, params flattened in insertion order)
    const int*           input_ids = (const int*)d_in[0];
    const float*         enc       = (const float*)d_in[1];
    const unsigned char* enc_mask  = (const unsigned char*)d_in[2];
    const int*           sub_mask  = (const int*)d_in[3];

    int p = 4;
    auto F = [&](int i) { return (const float*)d_in[i]; };
    auto getAttn = [&]() {
        AttnW a;
        a.Wq = F(p++); a.bq = F(p++); a.Wk = F(p++); a.bk = F(p++);
        a.Wv = F(p++); a.bv = F(p++); a.Wo = F(p++); a.bo = F(p++);
        return a;
    };
    const float* emb = F(p++);
    AttnW sa0 = getAttn();
    const float* ln_g = F(p++);
    const float* ln_b = F(p++);
    const float* intent_W = F(p++);
    const float* intent_b = F(p++);
    const float* slot_W   = F(p++);
    const float* slot_b   = F(p++);
    struct Layer { AttnW sa, ca; const float *W1,*b1,*W2,*b2,*n1g,*n1b,*n2g,*n2b,*n3g,*n3b; };
    Layer lay[2];
    for (int i = 0; i < 2; ++i) {
        lay[i].sa = getAttn();
        lay[i].ca = getAttn();
        lay[i].W1 = F(p++); lay[i].b1 = F(p++);
        lay[i].W2 = F(p++); lay[i].b2 = F(p++);
        lay[i].n1g = F(p++); lay[i].n1b = F(p++);
        lay[i].n2g = F(p++); lay[i].n2b = F(p++);
        lay[i].n3g = F(p++); lay[i].n3b = F(p++);
    }

    // ---- workspace carve
    const ll BLH  = (ll)BLk * Hk;             // 16.7M floats
    const ll SINT = (ll)Bk * NHSA * Lk * Lk;  // 33.5M floats (intent scores)
    float* buf0 = (float*)d_ws;
    float* buf1 = buf0 + BLH;
    float* buf2 = buf1 + BLH;
    float* buf3 = buf2 + BLH;
    float* buf4 = buf3 + BLH;
    float* sbuf = buf4 + BLH;
    float* pooled = sbuf + SINT;              // B*H floats

    float* out       = (float*)d_out;
    float* outSlot   = out;                          // [BL x SLOT]
    float* outIntent = out + (ll)BLk * SLOTk;        // [B x INTENT]

    const ll sLH = (ll)Lk * Hk;

    // ================= intent branch =================
    tok_gemm(stream, enc, sa0.Wq, sa0.bq, buf0, Hk, Hk, 0);   // Q
    tok_gemm(stream, enc, sa0.Wk, sa0.bk, buf1, Hk, Hk, 0);   // K
    tok_gemm(stream, enc, sa0.Wv, sa0.bv, buf2, Hk, Hk, 0);   // V
    // scores[b,h,q,k] = Q·K^T / sqrt(64); hd=64, nh=8
    launch_gemm(stream, buf0, buf1, nullptr, sbuf, Lk, Lk, 64,
                sLH, 64, Hk, sLH, 64, Hk, 1,
                (ll)NHSA * Lk * Lk, (ll)Lk * Lk, Lk,
                NHSA, Bk * NHSA, 0.125f, 0);
    softmax_kernel<<<Bk * NHSA * Lk, Lk, 0, stream>>>(sbuf, 0);
    // o[b,q,h,:] = a @ V   (strided back into (B,L,H) layout)
    launch_gemm(stream, sbuf, buf2, nullptr, buf3, Lk, 64, Lk,
                (ll)NHSA * Lk * Lk, (ll)Lk * Lk, Lk,
                sLH, 64, Hk, 0,
                sLH, 64, Hk, NHSA, Bk * NHSA, 1.0f, 0);
    tok_gemm(stream, buf3, sa0.Wo, sa0.bo, buf4, Hk, Hk, 0);  // ctx
    // enc2 = LN(ctx) + enc
    add_ln_kernel<<<BLk, 256, 0, stream>>>(buf0, buf4, enc, ln_g, ln_b, 1);
    meanpool_kernel<<<dim3(Bk, 2), 256, 0, stream>>>(buf0, enc_mask, pooled);
    launch_gemm(stream, pooled, intent_W, intent_b, outIntent, Bk, INTENTk, Hk,
                0, 0, Hk, 0, 0, INTENTk, 0, 0, 0, INTENTk, 1, 1, 1.0f, 0);

    // ================= subtoken compaction =================
    compact_kernel<<<Bk, 256, 0, stream>>>(enc, sub_mask, buf1);  // newtensor

    // ================= decoder =================
    embed_pe_kernel<<<(unsigned)((BLH + 255) / 256), 256, 0, stream>>>(input_ids, emb, buf2);

    for (int li = 0; li < 2; ++li) {
        const Layer& L = lay[li];
        // ---- causal self-attention (nh=2, hd=256)
        tok_gemm(stream, buf2, L.sa.Wq, L.sa.bq, buf0, Hk, Hk, 0);
        tok_gemm(stream, buf2, L.sa.Wk, L.sa.bk, buf3, Hk, Hk, 0);
        tok_gemm(stream, buf2, L.sa.Wv, L.sa.bv, buf4, Hk, Hk, 0);
        launch_gemm(stream, buf0, buf3, nullptr, sbuf, Lk, Lk, 256,
                    sLH, 256, Hk, sLH, 256, Hk, 1,
                    (ll)NHDEC * Lk * Lk, (ll)Lk * Lk, Lk,
                    NHDEC, Bk * NHDEC, 0.0625f, 0);
        softmax_kernel<<<Bk * NHDEC * Lk, Lk, 0, stream>>>(sbuf, 1);
        launch_gemm(stream, sbuf, buf4, nullptr, buf0, Lk, 256, Lk,
                    (ll)NHDEC * Lk * Lk, (ll)Lk * Lk, Lk,
                    sLH, 256, Hk, 0,
                    sLH, 256, Hk, NHDEC, Bk * NHDEC, 1.0f, 0);
        tok_gemm(stream, buf0, L.sa.Wo, L.sa.bo, buf3, Hk, Hk, 0);
        add_ln_kernel<<<BLk, 256, 0, stream>>>(buf2, buf3, buf2, L.n1g, L.n1b, 0);

        // ---- cross-attention: diag mask => output == V projection rows
        tok_gemm(stream, buf1, L.ca.Wv, L.ca.bv, buf0, Hk, Hk, 0);
        tok_gemm(stream, buf0, L.ca.Wo, L.ca.bo, buf3, Hk, Hk, 0);
        add_ln_kernel<<<BLk, 256, 0, stream>>>(buf2, buf3, buf2, L.n2g, L.n2b, 0);

        // ---- FFN
        tok_gemm(stream, buf2, L.W1, L.b1, buf0, FFk, Hk, 1);
        tok_gemm(stream, buf0, L.W2, L.b2, buf3, Hk, FFk, 0);
        add_ln_kernel<<<BLk, 256, 0, stream>>>(buf2, buf3, buf2, L.n3g, L.n3b, 0);
    }

    // ================= slot head =================
    tok_gemm(stream, buf2, slot_W, slot_b, outSlot, SLOTk, Hk, 0);
    log_softmax_kernel<<<BLk, 256, 0, stream>>>(outSlot, SLOTk);
}
